// ModulateConvBlock_88399016886416
// MI455X (gfx1250) — matrix operand order repaired and checked by reading, not compiled, not run on hardware
//
#include <hip/hip_runtime.h>

#define IN_C   64
#define OUT_C  64
#define RES    512
#define W_DIM  512
#define BATCH  4
#define ROWS   4             // output rows per block
#define COLS   64            // output cols per block
#define TILE_R (ROWS + 2)    // 6 (halo)
#define TILE_C (COLS + 2)    // 66 (halo)
#define CH_PAD 72            // padded per-pixel channel stride (shorts): 36 dwords,
                             // 9 coprime to 16 -> conflict-free ds_load_b128
#define PK_PER_B 18432       // dwords of packed A-fragments per batch: 9*2*4*32*8

typedef __attribute__((ext_vector_type(16))) __bf16 v16bf;
typedef __attribute__((ext_vector_type(8)))  float  v8f;

struct __align__(16) U128 { unsigned int w[4]; };
struct U256 { U128 lo, hi; };

__device__ __forceinline__ unsigned short f2bf(float f) {
  unsigned int u = __float_as_uint(f);
  unsigned int rnd = 0x7FFFu + ((u >> 16) & 1u);   // round-to-nearest-even
  return (unsigned short)((u + rnd) >> 16);
}

// 32B B-fragment load from LDS with only 16B alignment (padded stride)
__device__ __forceinline__ v16bf lds_frag(const unsigned short* p) {
  const U128* q = reinterpret_cast<const U128*>(p);
  U256 t; t.lo = q[0]; t.hi = q[2];
  return __builtin_bit_cast(v16bf, t);
}

// ---------------------------------------------------------------------------
// Kernel 1: style dense + modulate + demodulate, pack bf16 WMMA A-fragments
// (16x32 register layout per ISA 7.12.2).
// pk dword index: b*18432 + t*2048 + s*1024 + mg*256 + lane*8 + v
// ---------------------------------------------------------------------------
__global__ void prep_weights(const float* __restrict__ w,
                             const float* __restrict__ conv_w,
                             const float* __restrict__ style_w,
                             const float* __restrict__ style_b,
                             unsigned int* __restrict__ pk)
{
  const int b = blockIdx.x;
  const int tid = threadIdx.x;
  __shared__ float style[IN_C];
  __shared__ float invn[OUT_C];
  const float STYLE_WSCALE = 0.044194173824159216f;  // 1/sqrt(512)
  const float WSCALE       = 0.041666666666666664f;  // 1/sqrt(9*64)

  if (tid < IN_C) {
    float acc = 0.f;
    const float* wr = w + b * W_DIM;
    const float* sr = style_w + tid * W_DIM;
    for (int j = 0; j < W_DIM; ++j) acc += wr[j] * sr[j];
    style[tid] = acc * STYLE_WSCALE + style_b[tid] + 1.0f;
  }
  __syncthreads();

  if (tid < OUT_C) {
    float ss = 0.f;
    const float* cw = conv_w + tid * IN_C * 9;
    for (int k = 0; k < IN_C * 9; ++k) {
      float m = cw[k] * WSCALE * style[k / 9];
      ss += m * m;
    }
    invn[tid] = 1.0f / sqrtf(ss + 1e-8f);
  }
  __syncthreads();

  for (int idx = tid; idx < PK_PER_B; idx += blockDim.x) {
    const int v    = idx & 7;
    const int lane = (idx >> 3) & 31;
    const int mg   = (idx >> 8) & 3;
    const int s    = (idx >> 10) & 1;
    const int t    = idx >> 11;             // 0..8 tap
    const int kh = t / 3, kw = t - 3 * kh;
    const int M  = mg * 16 + (lane & 15);
    const int hi = lane >> 4;
    const int kbase = (v < 4) ? ((hi ? 8 : 0) + 2 * v)
                              : ((hi ? 24 : 16) + 2 * (v - 4));
    const int c0 = s * 32 + kbase;
    const float sc = WSCALE * invn[M];
    const float w0 = conv_w[((M * IN_C + c0    ) * 3 + kh) * 3 + kw] * sc * style[c0];
    const float w1 = conv_w[((M * IN_C + c0 + 1) * 3 + kh) * 3 + kw] * sc * style[c0 + 1];
    pk[b * PK_PER_B + idx] =
        (unsigned int)f2bf(w0) | ((unsigned int)f2bf(w1) << 16);
  }
}

// ---------------------------------------------------------------------------
// Kernel 2: implicit-GEMM 3x3 conv via v_wmma_f32_16x16x32_bf16.
// Block = 8 wave32s tiled 2(M) x 4(N): each wave owns 32 out-ch x 64 pixels
// (one output row) = 8 C tiles. Each B fragment feeds TWO WMMAs (A0/A1),
// halving LDS read traffic. B fragments for a K-step are batch-loaded into
// distinct registers so ds_load latency overlaps WMMA issue.
// ---------------------------------------------------------------------------
__global__ void __launch_bounds__(256)
mod_conv(const float* __restrict__ x,
         const unsigned int* __restrict__ pk,
         const float* __restrict__ conv_b,
         const float* __restrict__ noise,
         const float* __restrict__ noise_strength,
         float* __restrict__ out)
{
  __shared__ __align__(32) unsigned short tile[TILE_R * TILE_C * CH_PAD];

  const int tid     = threadIdx.x;
  const int b       = blockIdx.z;
  const int rowBase = blockIdx.y * ROWS;
  const int colBase = blockIdx.x * COLS;

  // ---- stage halo tile (zero padded) into LDS as bf16, channel pairs packed
  //      into one dword store; layout [row][col][channel(+pad)] ----
  {
    const int sub  = tid >> 6;   // 0..3 -> channel-pair sub-offset
    const int colT = tid & 63;   // coalesced along w
    const float* xb = x + (size_t)b * IN_C * RES * RES;
    for (int r = 0; r < TILE_R; ++r) {
      const int gh  = rowBase - 1 + r;
      const bool rok = (gh >= 0) && (gh < RES);
      for (int pb = 0; pb < 32; pb += 4) {
        const int c0 = 2 * (pb + sub);
        float v0 = 0.f, v1 = 0.f;
        if (rok) {
          const float* px = xb + (size_t)gh * RES + (colBase + colT);
          v0 = px[(size_t)c0 * RES * RES];
          v1 = px[(size_t)(c0 + 1) * RES * RES];
        }
        *reinterpret_cast<unsigned int*>(
            &tile[(r * TILE_C + 1 + colT) * CH_PAD + c0]) =
            (unsigned int)f2bf(v0) | ((unsigned int)f2bf(v1) << 16);
      }
      if (tid < 64) {            // halo columns 0 and 65
        const int side = tid & 1;
        const int c0   = 2 * (tid >> 1);
        const int gw   = side ? (colBase + COLS) : (colBase - 1);
        float v0 = 0.f, v1 = 0.f;
        if (rok && gw >= 0 && gw < RES) {
          const float* px = xb + (size_t)gh * RES + gw;
          v0 = px[(size_t)c0 * RES * RES];
          v1 = px[(size_t)(c0 + 1) * RES * RES];
        }
        *reinterpret_cast<unsigned int*>(
            &tile[(r * TILE_C + (side ? TILE_C - 1 : 0)) * CH_PAD + c0]) =
            (unsigned int)f2bf(v0) | ((unsigned int)f2bf(v1) << 16);
      }
    }
  }
  __syncthreads();

  const int wv   = tid >> 5;
  const int lane = tid & 31;
  const int mh   = wv & 1;      // M half: 32 out channels (2 A fragments)
  const int nq   = wv >> 1;     // 0..3 : output row within block
  const int n    = lane & 15;
  const int half = lane >> 4;

  v8f acc[2][4];
#pragma unroll
  for (int a = 0; a < 2; ++a)
#pragma unroll
    for (int i = 0; i < 4; ++i) acc[a][i] = v8f{};

  const unsigned int* pk0 = pk + b * PK_PER_B + (2 * mh) * 256 + lane * 8;

  for (int kh = 0; kh < 3; ++kh) {
    for (int kw = 0; kw < 3; ++kw) {
      const int t = kh * 3 + kw;
      const int rowOff = ((nq + kh) * TILE_C + kw + n) * CH_PAD;
#pragma unroll
      for (int s = 0; s < 2; ++s) {
        const unsigned int* ap = pk0 + t * 2048 + s * 1024;
        const v16bf A0 = *reinterpret_cast<const v16bf*>(ap);
        const v16bf A1 = *reinterpret_cast<const v16bf*>(ap + 256);
        const int ch = s * 32 + half * 8;
        v16bf Bf[4];
#pragma unroll
        for (int nt = 0; nt < 4; ++nt)
          Bf[nt] = lds_frag(&tile[rowOff + nt * 16 * CH_PAD + ch]);
#pragma unroll
        for (int nt = 0; nt < 4; ++nt) {
          acc[0][nt] = __builtin_amdgcn_wmma_f32_16x16x32_bf16(
              false, A0, false, Bf[nt], (short)0, acc[0][nt], false, false);
          acc[1][nt] = __builtin_amdgcn_wmma_f32_16x16x32_bf16(
              false, A1, false, Bf[nt], (short)0, acc[1][nt], false, false);
        }
      }
    }
  }

  // ---- fused epilogue: noise + bias + lrelu*sqrt(2) ----
  const float ns = noise_strength[0];
  const int gh = rowBase + nq;
#pragma unroll
  for (int nt = 0; nt < 4; ++nt) {
    const int gw = colBase + nt * 16 + n;
    const float nz = noise[gh * RES + gw] * ns;
#pragma unroll
    for (int a = 0; a < 2; ++a) {
#pragma unroll
      for (int j = 0; j < 8; ++j) {
        const int o = (2 * mh + a) * 16 + half * 8 + j;  // C layout: VGPR j -> M=j(+8)
        float v = acc[a][nt][j] + nz + conv_b[o];
        v = (v >= 0.f ? v : 0.2f * v) * 1.4142135623730951f;
        out[(((size_t)b * OUT_C + o) * RES + gh) * RES + gw] = v;
      }
    }
  }
}

extern "C" void kernel_launch(void* const* d_in, const int* in_sizes, int n_in,
                              void* d_out, int out_size, void* d_ws, size_t ws_size,
                              hipStream_t stream) {
  const float* x       = (const float*)d_in[0];
  const float* w       = (const float*)d_in[1];
  const float* conv_w  = (const float*)d_in[2];
  const float* style_w = (const float*)d_in[3];
  const float* style_b = (const float*)d_in[4];
  const float* conv_b  = (const float*)d_in[5];
  const float* noise   = (const float*)d_in[6];
  const float* nstr    = (const float*)d_in[7];
  unsigned int* pk = (unsigned int*)d_ws;   // 4 * 18432 dwords = 288 KB

  prep_weights<<<BATCH, 256, 0, stream>>>(w, conv_w, style_w, style_b, pk);

  dim3 grid(RES / COLS, RES / ROWS, BATCH);   // 8 x 128 x 4 blocks
  mod_conv<<<grid, 256, 0, stream>>>(x, pk, conv_b, noise, nstr, (float*)d_out);
}
